// SVGDModel_86955907875267
// MI455X (gfx1250) — matrix operand order repaired and checked by reading, not compiled, not run on hardware
//
#include <hip/hip_runtime.h>
#include <hip/hip_fp16.h>
#include <stdint.h>

typedef _Float16 half16 __attribute__((ext_vector_type(16)));
typedef float    float8 __attribute__((ext_vector_type(8)));

#define N_PART 4096
#define DIMS   32

// half-wave swap (lane ^ 16) via ds_swizzle group-of-32: xor=0x10, or=0, and=0x1f
__device__ __forceinline__ float swapx16(float x) {
    return __int_as_float(__builtin_amdgcn_ds_swizzle(__float_as_int(x), 0x401f));
}

// A-operand K index base for 16-bit 16x32 A layout (lane half 'hi', VGPR v)
__device__ __forceinline__ int a_kbase(int v, int hi) {
    return ((v < 4) ? 0 : 16) + (hi ? 8 : 0) + 2 * (v & 3);
}

// ---------------------------------------------------------------------------
// Pass 1: X(f32) -> Xh (f16 [N][32]), XhT (f16 [32][N]), Nsq[j] = ||x_j||^2
// ---------------------------------------------------------------------------
__global__ void svgd_pre(const float* __restrict__ X,
                         _Float16* __restrict__ Xh,
                         _Float16* __restrict__ XhT,
                         float* __restrict__ Nsq) {
    int j = blockIdx.x * blockDim.x + threadIdx.x;
    if (j >= N_PART) return;
    float acc = 0.f;
#pragma unroll
    for (int d = 0; d < DIMS; ++d) {
        float v = X[j * DIMS + d];
        acc += v * v;
        _Float16 h = (_Float16)v;
        Xh[j * DIMS + d]    = h;
        XhT[d * N_PART + j] = h;
    }
    Nsq[j] = acc;
}

// ---------------------------------------------------------------------------
// Pass 2: fused SVGD. One block = one 16-particle i-tile, 4 waves split j.
// Per 32-j chunk (all register dataflow, no LDS in the loop):
//   Gram (M=j, N=i):  g_s = Xj_s (A) x Xi (B)           -> 2 WMMA
//   K = exp2(c * sqdist) on VALU
//   C-layout -> B-layout relayout via ds_swizzle SWAPX16 + cndmask + f16 pack
//   OUT^T[d,i] += XhT (A) x K (B)                        -> 2 WMMA
//   rowsum s[i] += ones (A) x K (B)                      -> 1 WMMA
// ---------------------------------------------------------------------------
__global__ void __launch_bounds__(128) svgd_main(const float* __restrict__ X,
                                                 const _Float16* __restrict__ Xh,
                                                 const _Float16* __restrict__ XhT,
                                                 const float* __restrict__ Nsq,
                                                 const float* __restrict__ ls,
                                                 float* __restrict__ out) {
    __shared__ float red[4][32][17];     // cross-wave reduction buffer (8.5 KB)

    const int lane = threadIdx.x & 31;
    const int wave = threadIdx.x >> 5;
    const int i0   = blockIdx.x * 16;
    const int hi   = lane >> 4;          // half-wave select
    const int lm   = lane & 15;
    const bool hb  = (hi != 0);

    const float l2    = ls[0] * ls[0];
    const float invl2 = 1.0f / l2;
    const float c     = -1.44269504088896340736f * (0.5f * invl2);  // exp2 scale

    // B operand of the Gram WMMA: Xi tile, B[k=d][n=i] = X[i0+n][k].
    // B layout: lane -> N = lane&15; VGPR v holds K pair {a_kbase(v, hi), +1}.
    union { half16 h; uint32_t u[8]; } bXi;
#pragma unroll
    for (int v = 0; v < 8; ++v)
        bXi.u[v] = *(const uint32_t*)(Xh + (i0 + lm) * DIMS + a_kbase(v, hi));

    const float niv = Nsq[i0 + lm];      // ||x_i||^2 for this lane's column

    half16 aones;                         // all-ones A -> rowsum on matrix pipe
#pragma unroll
    for (int k = 0; k < 16; ++k) aones[k] = (_Float16)1.0f;

    float8 kx0 = {}, kx1 = {}, sac = {};

    for (int it = 0; it < N_PART / 128; ++it) {     // uniform trip count (32)
        const int j0 = it * 128 + wave * 32;

        // Gram A operands: A[m=jr][k=d] = X[js+jr][d], js = j0 / j0+16.
        union { half16 h; uint32_t u[8]; } aJ0, aJ1;
#pragma unroll
        for (int v = 0; v < 8; ++v) {
            int kb = a_kbase(v, hi);
            aJ0.u[v] = *(const uint32_t*)(Xh + (j0 + lm) * DIMS + kb);
            aJ1.u[v] = *(const uint32_t*)(Xh + (j0 + 16 + lm) * DIMS + kb);
        }
        float8 z = {};
        float8 g0 = __builtin_amdgcn_wmma_f32_16x16x32_f16(false, aJ0.h, false, bXi.h,
                                                           (short)0, z, false, false);
        float8 g1 = __builtin_amdgcn_wmma_f32_16x16x32_f16(false, aJ1.h, false, bXi.h,
                                                           (short)0, z, false, false);

        // C layout: lane,vgpr r -> (jr = r + 8*hi, i = lm).
        // K[i0+lm, j0 + r + 8*hi] and K[i0+lm, j0+16 + r + 8*hi]:
        float kv0[8], kv1[8];
#pragma unroll
        for (int r = 0; r < 8; ++r) {
            float nj0 = Nsq[j0 + r + 8 * hi];
            float nj1 = Nsq[j0 + 16 + r + 8 * hi];
            kv0[r] = __builtin_amdgcn_exp2f((nj0 + niv - 2.0f * g0[r]) * c);
            kv1[r] = __builtin_amdgcn_exp2f((nj1 + niv - 2.0f * g1[r]) * c);
        }
        // Half-wave swapped copies: sw*[r] = K[i0+lm, j0(+16) + r + 8*(!hi)]
        float sw0[8], sw1[8];
#pragma unroll
        for (int r = 0; r < 8; ++r) { sw0[r] = swapx16(kv0[r]); sw1[r] = swapx16(kv1[r]); }

        // Assemble B operand b[k=j-in-chunk][n=i]: lane -> N=lm,
        // VGPR v holds K pair {16*hi + 2v, +1}.
        union { half16 h; uint32_t u[8]; } b;
#pragma unroll
        for (int v = 0; v < 4; ++v) {
            float lo = hb ? sw1[2 * v]     : kv0[2 * v];
            float he = hb ? sw1[2 * v + 1] : kv0[2 * v + 1];
            b.h[2 * v]     = (_Float16)lo;
            b.h[2 * v + 1] = (_Float16)he;
        }
#pragma unroll
        for (int v = 4; v < 8; ++v) {
            int r2 = 2 * v - 8;
            float lo = hb ? kv1[r2]     : sw0[r2];
            float he = hb ? kv1[r2 + 1] : sw0[r2 + 1];
            b.h[2 * v]     = (_Float16)lo;
            b.h[2 * v + 1] = (_Float16)he;
        }

        // KX^T A operands: A[m=d'][k=jr] = XhT[d' + 16*half][j0 + jr].
        union { half16 h; uint32_t u[8]; } aX0, aX1;
#pragma unroll
        for (int v = 0; v < 8; ++v) {
            int kb = 16 * hi + 2 * v;
            aX0.u[v] = *(const uint32_t*)(XhT + lm        * N_PART + j0 + kb);
            aX1.u[v] = *(const uint32_t*)(XhT + (16 + lm) * N_PART + j0 + kb);
        }

        kx0 = __builtin_amdgcn_wmma_f32_16x16x32_f16(false, aX0.h, false, b.h,
                                                     (short)0, kx0, false, false);
        kx1 = __builtin_amdgcn_wmma_f32_16x16x32_f16(false, aX1.h, false, b.h,
                                                     (short)0, kx1, false, false);
        sac = __builtin_amdgcn_wmma_f32_16x16x32_f16(false, aones, false, b.h,
                                                     (short)0, sac, false, false);
    }

    // Cross-wave reduction (sac is m-replicated: one value per lane suffices).
#pragma unroll
    for (int r = 0; r < 8; ++r) {
        red[wave][lane][r]     = kx0[r];
        red[wave][lane][8 + r] = kx1[r];
    }
    red[wave][lane][16] = sac[0];
    __syncthreads();

    if (wave == 0) {
        const float scale = 1.0f / (float)N_PART;
        const float s = red[0][lane][16] + red[1][lane][16] +
                        red[2][lane][16] + red[3][lane][16];
        const int row = i0 + lm;
#pragma unroll
        for (int r = 0; r < 8; ++r) {
            float a0 = red[0][lane][r]     + red[1][lane][r]     + red[2][lane][r]     + red[3][lane][r];
            float a1 = red[0][lane][8 + r] + red[1][lane][8 + r] + red[2][lane][8 + r] + red[3][lane][8 + r];
            int d0 = r + 8 * hi;
            int d1 = 16 + r + 8 * hi;
            float x0 = X[row * DIMS + d0];
            float x1 = X[row * DIMS + d1];
            // phi = (x*s/l^2 - (1 + 1/l^2) * (K@X)) / N
            out[row * DIMS + d0] = (x0 * s * invl2 - (1.0f + invl2) * a0) * scale;
            out[row * DIMS + d1] = (x1 * s * invl2 - (1.0f + invl2) * a1) * scale;
        }
    }
}

// ---------------------------------------------------------------------------
extern "C" void kernel_launch(void* const* d_in, const int* in_sizes, int n_in,
                              void* d_out, int out_size, void* d_ws, size_t ws_size,
                              hipStream_t stream) {
    (void)in_sizes; (void)n_in; (void)out_size; (void)ws_size;

    const float* X  = (const float*)d_in[0];   // [4096,32] f32
    const float* ls = (const float*)d_in[1];   // [1] f32
    float* out = (float*)d_out;                // [4096,32] f32

    char* ws = (char*)d_ws;
    _Float16* Xh  = (_Float16*)(ws);                         // 256 KB
    _Float16* XhT = (_Float16*)(ws + N_PART * DIMS * 2);     // 256 KB
    float*    Nsq = (float*)   (ws + 2 * N_PART * DIMS * 2); // 16 KB

    svgd_pre<<<N_PART / 256, 256, 0, stream>>>(X, Xh, XhT, Nsq);
    svgd_main<<<N_PART / 16, 128, 0, stream>>>(X, Xh, XhT, Nsq, ls, out);
}
